// DeformVideo2D_with_FPN_61246233641572
// MI455X (gfx1250) — matrix operand order repaired and checked by reading, not compiled, not run on hardware
//
#include <hip/hip_runtime.h>

typedef __attribute__((ext_vector_type(16))) __bf16 v16bf;
typedef __attribute__((ext_vector_type(8)))  float  v8f;
typedef __attribute__((ext_vector_type(4)))  unsigned int u32x4;
typedef unsigned short u16;
typedef unsigned int   u32;

#define BT_   8
#define LEN_  3024
#define D_    256
#define M_    (BT_*LEN_)      // 24192 tokens
#define NH_   8
#define HD_   32
#define NL_   3
#define NP_   4
#define DFF_  2048

__device__ __forceinline__ u16 f2bf(float f) {
    u32 u = __builtin_bit_cast(u32, f);
    u32 r = u + 0x7FFFu + ((u >> 16) & 1u);   // round-to-nearest-even
    return (u16)(r >> 16);
}

union FragBF { v16bf v; u32 u[8]; u32x4 q[2]; };

// A-fragment (16x32 bf16, ISA 7.12.2): per-lane dwords {0..3}+half*4 and {8..11}+half*4
__device__ __forceinline__ FragBF load_frag_A(const u32* __restrict__ dbase, int half) {
    FragBF f;
    f.q[0] = *(const u32x4*)(dbase + half * 4);
    f.q[1] = *(const u32x4*)(dbase + half * 4 + 8);
    return f;
}
// B-fragment (32x16 bf16): per-lane dwords {0..3}+half*4 and {8..11}+half*4 (column = lane&15, N-major storage)
__device__ __forceinline__ FragBF load_frag_B(const u32* __restrict__ dbase, int half) {
    FragBF f;
    f.q[0] = *(const u32x4*)(dbase + half * 4);
    f.q[1] = *(const u32x4*)(dbase + half * 4 + 8);
    return f;
}

// ---------------------------------------------------------------------------
// GEMM (K compile-time): C[M][N] = A_bf16[M][K] @ Wt_bf16[N][K] + bias.
// One wave computes a 16x(16*NT) strip.  All K/32 A-fragments preloaded into
// registers once; only B streams (8 b128 loads per 4 WMMAs at NT=4).
// ---------------------------------------------------------------------------
template <int NT, int K, bool RELU>
__global__ __launch_bounds__(128, 1)
void gemm_bf16_kernel(const u16* __restrict__ A, const u16* __restrict__ Wt,
                      const float* __restrict__ bias, float* __restrict__ Cf, int N) {
    constexpr int KT = K / 32;
    int lane = threadIdx.x & 31, wid = threadIdx.x >> 5;
    int half = lane >> 4, l16 = lane & 15;
    int mt = blockIdx.x;
    int ng = blockIdx.y * 4 + wid;
    int n0 = ng * (16 * NT);
    if (n0 >= N) return;
    int m0 = mt * 16;
    const u32* A32 = (const u32*)A;
    const u32* W32 = (const u32*)Wt;
    const u32* arow = A32 + (((m0 + l16) * K) >> 1);
    const u32* brow[NT];
#pragma unroll
    for (int j = 0; j < NT; j++) brow[j] = W32 + (((n0 + j * 16 + l16) * K) >> 1);

    FragBF afrag[KT];
#pragma unroll
    for (int t = 0; t < KT; t++) afrag[t] = load_frag_A(arow + t * 16, half);

    v8f zero = {};
    v8f acc[NT];
#pragma unroll
    for (int j = 0; j < NT; j++) acc[j] = zero;

#pragma unroll
    for (int t = 0; t < KT; t++) {
        FragBF b[NT];
#pragma unroll
        for (int j = 0; j < NT; j++) b[j] = load_frag_B(brow[j] + t * 16, half);
#pragma unroll
        for (int j = 0; j < NT; j++)
            acc[j] = __builtin_amdgcn_wmma_f32_16x16x32_bf16(false, afrag[t].v, false, b[j].v,
                                                             (short)0, acc[j], false, false);
    }
#pragma unroll
    for (int j = 0; j < NT; j++) {
        int n = n0 + j * 16 + l16;
        float bs = bias[n];
#pragma unroll
        for (int v = 0; v < 8; v++) {
            int m = m0 + v + 8 * half;
            float r = acc[j][v] + bs;
            if (RELU) r = fmaxf(r, 0.f);
            Cf[(long)m * N + n] = r;
        }
    }
}

// ---------------------------------------------------------------------------
// Fused FFN: y = relu(x@W1+b1)@W2+b2.  Block = 8 waves / 16 token rows; the
// 16x2048 bf16 hidden tile lives in 64KB LDS.  A-fragments (8) are loaded
// once per block and reused across all four 64-column groups (no respill).
// ---------------------------------------------------------------------------
__global__ __launch_bounds__(256, 1)
void ffn_kernel(const u16* __restrict__ Xb, const u16* __restrict__ W1t,
                const float* __restrict__ b1, const u16* __restrict__ W2t,
                const float* __restrict__ b2, float* __restrict__ Y) {
    __shared__ u32 hbuf[16 * (DFF_ / 2)];   // 16 x 2048 bf16 = 64KB
    int lane = threadIdx.x & 31, wid = threadIdx.x >> 5;
    int half = lane >> 4, l16 = lane & 15;
    int m0 = blockIdx.x * 16;
    const u32* A32 = (const u32*)Xb;
    const u32* W1_32 = (const u32*)W1t;
    const u32* arow = A32 + (((m0 + l16) * D_) >> 1);
    v8f zero = {};
    u16* hs = (u16*)hbuf;

    FragBF afrag[8];
#pragma unroll
    for (int t = 0; t < 8; t++) afrag[t] = load_frag_A(arow + t * 16, half);

    // phase 1: hidden = relu(x @ W1 + b1) -> LDS, 64 columns per wave-iteration
    for (int g = wid; g < DFF_ / 64; g += 8) {
        const u32* brow[4];
#pragma unroll
        for (int j = 0; j < 4; j++) brow[j] = W1_32 + (((g * 64 + j * 16 + l16) * D_) >> 1);
        v8f acc[4];
#pragma unroll
        for (int j = 0; j < 4; j++) acc[j] = zero;
#pragma unroll
        for (int t = 0; t < 8; t++) {
            FragBF b[4];
#pragma unroll
            for (int j = 0; j < 4; j++) b[j] = load_frag_B(brow[j] + t * 16, half);
#pragma unroll
            for (int j = 0; j < 4; j++)
                acc[j] = __builtin_amdgcn_wmma_f32_16x16x32_bf16(false, afrag[t].v, false, b[j].v,
                                                                 (short)0, acc[j], false, false);
        }
#pragma unroll
        for (int j = 0; j < 4; j++) {
            int n = g * 64 + j * 16 + l16;
            float bs = b1[n];
#pragma unroll
            for (int v = 0; v < 8; v++) {
                int m = v + 8 * half;
                hs[m * DFF_ + n] = f2bf(fmaxf(acc[j][v] + bs, 0.f));
            }
        }
    }
    __syncthreads();

    // phase 2: y = hidden @ W2 + b2; 8 waves x 32 columns (NT=2), A from LDS b128
    const u32* W2_32 = (const u32*)W2t;
    const u32* arow2 = hbuf + ((l16 * DFF_) >> 1);
    {
        int n0 = wid * 32;
        const u32* brow[2];
#pragma unroll
        for (int j = 0; j < 2; j++) brow[j] = W2_32 + (((n0 + j * 16 + l16) * DFF_) >> 1);
        v8f acc[2];
        acc[0] = zero; acc[1] = zero;
        for (int k0 = 0; k0 < DFF_; k0 += 32) {
            int kd = k0 >> 1;
            FragBF a = load_frag_A(arow2 + kd, half);
            FragBF b0 = load_frag_B(brow[0] + kd, half);
            FragBF b1f = load_frag_B(brow[1] + kd, half);
            acc[0] = __builtin_amdgcn_wmma_f32_16x16x32_bf16(false, a.v, false, b0.v,
                                                             (short)0, acc[0], false, false);
            acc[1] = __builtin_amdgcn_wmma_f32_16x16x32_bf16(false, a.v, false, b1f.v,
                                                             (short)0, acc[1], false, false);
        }
#pragma unroll
        for (int j = 0; j < 2; j++) {
            int n = n0 + j * 16 + l16;
            float bs = b2[n];
#pragma unroll
            for (int v = 0; v < 8; v++) {
                int m = m0 + v + 8 * half;
                Y[(long)m * D_ + n] = acc[j][v] + bs;
            }
        }
    }
}

// ---------------------------------------------------------------------------
// 3x3 conv as WMMA GEMM (K = 9*256; every 32-wide K chunk is exactly one tap).
// Zero padding via per-lane vector select (EXEC stays all-ones for WMMA).
// Output written CHW into d_out; groupnorm runs in place afterwards.
// ---------------------------------------------------------------------------
__global__ __launch_bounds__(128, 1)
void conv3_kernel(const u16* __restrict__ FB, const u16* __restrict__ Wt,
                  const float* __restrict__ bias, float* __restrict__ O) {
    int lane = threadIdx.x & 31, wid = threadIdx.x >> 5;
    int half = lane >> 4, l16 = lane & 15;
    int mt = blockIdx.x;                 // bt*576 + y*6 + xg
    int xg = mt % 6, y = (mt / 6) % 96, bt = mt / (6 * 96);
    int x0 = xg * 16, n0 = wid * 64;     // each wave: 64 output channels
    const u32* A32 = (const u32*)FB;
    const u32* W32 = (const u32*)Wt;
    const u32* brow[4];
#pragma unroll
    for (int j = 0; j < 4; j++) brow[j] = W32 + (((n0 + j * 16 + l16) * (9 * D_)) >> 1);
    v8f zero = {};
    v8f acc[4];
#pragma unroll
    for (int j = 0; j < 4; j++) acc[j] = zero;
    u32x4 zq = {0u, 0u, 0u, 0u};

    auto loadA = [&](int kt) -> FragBF {
        int tap = kt >> 3, ci0 = (kt & 7) * 32;
        int dy = tap / 3 - 1, dx = tap % 3 - 1;
        int py = y + dy, px = x0 + l16 + dx;
        bool ok = (py >= 0) && (py < 96) && (px >= 0) && (px < 96);
        int dbase = ok ? (((bt * 96 + py) * 96 + px) * (D_ / 2) + (ci0 >> 1)) : 0;
        FragBF a;
        a.q[0] = *(const u32x4*)(A32 + dbase + half * 4);
        a.q[1] = *(const u32x4*)(A32 + dbase + half * 4 + 8);
        if (!ok) { a.q[0] = zq; a.q[1] = zq; }
        return a;
    };

    for (int kt = 0; kt < 72; kt++) {
        FragBF a = loadA(kt);
        FragBF b[4];
        int kd = kt * 16;
#pragma unroll
        for (int j = 0; j < 4; j++) b[j] = load_frag_B(brow[j] + kd, half);
#pragma unroll
        for (int j = 0; j < 4; j++)
            acc[j] = __builtin_amdgcn_wmma_f32_16x16x32_bf16(false, a.v, false, b[j].v,
                                                             (short)0, acc[j], false, false);
    }
#pragma unroll
    for (int j = 0; j < 4; j++) {
        int c = n0 + j * 16 + l16;
        float bs = bias[c];
#pragma unroll
        for (int v = 0; v < 8; v++) {
            int m = v + 8 * half;
            int px = x0 + m;
            O[((long)(bt * D_) + c) * 9216 + y * 96 + px] = acc[j][v] + bs;
        }
    }
}

// ---------------------------------------------------------------------------
// Pack feats/pos (CHW per level) into token-major x/pos + bf16 mirrors + mask.
// ---------------------------------------------------------------------------
__global__ void pack_kernel(const float* f8, const float* f16, const float* f32_,
                            const float* p8, const float* p16, const float* p32_,
                            const unsigned char* m8, const unsigned char* m16,
                            const unsigned char* m32_, const float* lvlemb,
                            float* X, float* P, u16* Xb, u16* Qb, unsigned char* Mf) {
    int idx = blockIdx.x * blockDim.x + threadIdx.x;
    if (idx >= M_ * D_) return;
    int d = idx & 255, tok = (idx >> 8) % LEN_, bt = idx / (LEN_ * D_);
    const float *f, *p; int s, hw, lvl;
    if (tok < 2304)      { lvl = 0; s = tok;        hw = 2304; f = f8;   p = p8;   }
    else if (tok < 2880) { lvl = 1; s = tok - 2304; hw = 576;  f = f16;  p = p16;  }
    else                 { lvl = 2; s = tok - 2880; hw = 144;  f = f32_; p = p32_; }
    float xv = f[((long)(bt * D_) + d) * hw + s];
    float pv = p[((long)(bt * D_) + d) * hw + s] + lvlemb[lvl * D_ + d];
    X[idx] = xv; P[idx] = pv;
    Xb[idx] = f2bf(xv); Qb[idx] = f2bf(xv + pv);
    if (d == 0) {
        const unsigned char* mm = (lvl == 0) ? m8 : ((lvl == 1) ? m16 : m32_);
        Mf[bt * LEN_ + tok] = mm[bt * hw + s];
    }
}

__global__ void vr_kernel(const unsigned char* m8, const unsigned char* m16,
                          const unsigned char* m32_, float* VR) {
    int i = blockIdx.x * blockDim.x + threadIdx.x;
    if (i >= BT_ * NL_) return;
    int bt = i / 3, lvl = i % 3;
    int H = (lvl == 0) ? 48 : ((lvl == 1) ? 24 : 12);
    const unsigned char* mm = (lvl == 0) ? m8 : ((lvl == 1) ? m16 : m32_);
    const unsigned char* base = mm + bt * H * H;
    int ch = 0, cw = 0;
    for (int r = 0; r < H; r++) ch += base[r * H] ? 0 : 1;
    for (int c = 0; c < H; c++) cw += base[c] ? 0 : 1;
    VR[i * 2 + 0] = (float)cw / H;
    VR[i * 2 + 1] = (float)ch / H;
}

__global__ void ref_kernel(const float* VR, float* REF) {
    int i = blockIdx.x * blockDim.x + threadIdx.x;
    if (i >= BT_ * LEN_ * NL_) return;
    int lvl2 = i % 3, tok = (i / 3) % LEN_, bt = i / (3 * LEN_);
    int slvl, s, H;
    if (tok < 2304)      { slvl = 0; s = tok;        H = 48; }
    else if (tok < 2880) { slvl = 1; s = tok - 2304; H = 24; }
    else                 { slvl = 2; s = tok - 2880; H = 12; }
    int iy = s / H, ix = s % H;
    float rx = (ix + 0.5f) / (VR[(bt * 3 + slvl) * 2 + 0] * H);
    float ry = (iy + 0.5f) / (VR[(bt * 3 + slvl) * 2 + 1] * H);
    REF[i * 2 + 0] = rx * VR[(bt * 3 + lvl2) * 2 + 0];
    REF[i * 2 + 1] = ry * VR[(bt * 3 + lvl2) * 2 + 1];
}

__global__ void mask_value_kernel(float* V, const unsigned char* Mf) {
    int idx = blockIdx.x * blockDim.x + threadIdx.x;
    if (idx >= M_ * D_) return;
    if (Mf[idx >> 8]) V[idx] = 0.f;
}

__global__ void softmax12_kernel(float* A) {
    int i = blockIdx.x * blockDim.x + threadIdx.x;
    if (i >= M_ * NH_) return;
    float* p = A + (long)i * 12;
    float mx = p[0];
    for (int j = 1; j < 12; j++) mx = fmaxf(mx, p[j]);
    float e[12], s = 0.f;
    for (int j = 0; j < 12; j++) { e[j] = __expf(p[j] - mx); s += e[j]; }
    float inv = 1.f / s;
    for (int j = 0; j < 12; j++) p[j] = e[j] * inv;
}

// One wave per (token, head); lane = channel (HD==32==wave32). Branchless
// bilinear: clamped indices, validity folded into the corner weights.
__global__ void ms_sample_kernel(const float* __restrict__ V, const float* __restrict__ OFF,
                                 const float* __restrict__ AT, const float* __restrict__ REF,
                                 u16* __restrict__ MS) {
    int wave = (blockIdx.x * blockDim.x + threadIdx.x) >> 5;
    int d = threadIdx.x & 31;
    if (wave >= M_ * NH_) return;
    int h = wave & 7, tq = wave >> 3;
    int bt = tq / LEN_;
    const float* offp = OFF + ((long)tq * NH_ + h) * 24;
    const float* atp  = AT  + ((long)tq * NH_ + h) * 12;
    const float* refp = REF + (long)tq * 6;
    const int st[3] = {0, 2304, 2880};
    const int HS[3] = {48, 24, 12};
    float acc = 0.f;
#pragma unroll
    for (int lvl = 0; lvl < 3; lvl++) {
        int H = HS[lvl], W = H;
        float rfx = refp[lvl * 2 + 0], rfy = refp[lvl * 2 + 1];
        const float* vbase = V + ((long)(bt * LEN_ + st[lvl])) * D_ + h * HD_ + d;
#pragma unroll
        for (int p = 0; p < 4; p++) {
            float a  = atp[lvl * 4 + p];
            float lx = rfx + offp[(lvl * 4 + p) * 2 + 0] / (float)W;
            float ly = rfy + offp[(lvl * 4 + p) * 2 + 1] / (float)H;
            float x = lx * W - 0.5f, y = ly * H - 0.5f;
            float x0f = floorf(x), y0f = floorf(y);
            float wx = x - x0f, wy = y - y0f;
            int x0 = (int)x0f, y0 = (int)y0f;
            float vx0 = (x0 >= 0 && x0 < W) ? 1.f : 0.f;
            float vx1 = (x0 + 1 >= 0 && x0 + 1 < W) ? 1.f : 0.f;
            float vy0 = (y0 >= 0 && y0 < H) ? 1.f : 0.f;
            float vy1 = (y0 + 1 >= 0 && y0 + 1 < H) ? 1.f : 0.f;
            int x0c = min(max(x0, 0), W - 1), x1c = min(max(x0 + 1, 0), W - 1);
            int y0c = min(max(y0, 0), H - 1), y1c = min(max(y0 + 1, 0), H - 1);
            float s00 = vbase[(long)(y0c * W + x0c) * D_];
            float s10 = vbase[(long)(y0c * W + x1c) * D_];
            float s01 = vbase[(long)(y1c * W + x0c) * D_];
            float s11 = vbase[(long)(y1c * W + x1c) * D_];
            acc += a * (s00 * ((1 - wx) * (1 - wy) * vx0 * vy0)
                      + s10 * (wx * (1 - wy) * vx1 * vy0)
                      + s01 * ((1 - wx) * wy * vx0 * vy1)
                      + s11 * (wx * wy * vx1 * vy1));
        }
    }
    MS[(long)tq * D_ + h * HD_ + d] = f2bf(acc);
}

// x = LN(x + delta); refresh bf16 mirrors (and q = bf16(x + pos) if requested)
__global__ void add_ln_kernel(float* __restrict__ X, const float* __restrict__ Dlt,
                              const float* __restrict__ g, const float* __restrict__ bta,
                              u16* __restrict__ Xb, const float* __restrict__ Pos,
                              u16* __restrict__ Qb) {
    __shared__ float red[256];
    int t = threadIdx.x;
    long base = (long)blockIdx.x * D_ + t;
    float v = X[base] + Dlt[base];
    red[t] = v; __syncthreads();
    for (int s = 128; s > 0; s >>= 1) { if (t < s) red[t] += red[t + s]; __syncthreads(); }
    float mean = red[0] / (float)D_; __syncthreads();
    float dd = v - mean;
    red[t] = dd * dd; __syncthreads();
    for (int s = 128; s > 0; s >>= 1) { if (t < s) red[t] += red[t + s]; __syncthreads(); }
    float var = red[0] / (float)D_;
    float y = dd * rsqrtf(var + 1e-5f) * g[t] + bta[t];
    X[base] = y;
    Xb[base] = f2bf(y);
    if (Qb) Qb[base] = f2bf(y + Pos[base]);
}

// Write mems (token-major x -> B,T,D,h,w) into d_out after the s4 block.
__global__ void unflatten_kernel(const float* __restrict__ X, float* __restrict__ OUT) {
    int idx = blockIdx.x * blockDim.x + threadIdx.x;
    if (idx >= M_ * D_) return;
    int d = idx & 255, tok = (idx >> 8) % LEN_, bt = idx / (LEN_ * D_);
    const long o0 = (long)BT_ * D_ * 9216;
    long base; int s, hw;
    if (tok < 2304)      { s = tok;        hw = 2304; base = o0; }
    else if (tok < 2880) { s = tok - 2304; hw = 576;  base = o0 + (long)BT_ * D_ * 2304; }
    else                 { s = tok - 2880; hw = 144;  base = o0 + (long)BT_ * D_ * 2304 + (long)BT_ * D_ * 576; }
    OUT[base + ((long)(bt * D_) + d) * hw + s] = X[idx];
}

// weight converters ---------------------------------------------------------
__global__ void wt_transpose_kernel(const float* W, u16* Wt, int K, int N) {
    int i = blockIdx.x * blockDim.x + threadIdx.x;
    if (i >= K * N) return;
    int n = i % N, k = i / N;
    Wt[(long)n * K + k] = f2bf(W[i]);
}
__global__ void cvt_kernel(const float* S, u16* Dd, int n) {
    int i = blockIdx.x * blockDim.x + threadIdx.x;
    if (i < n) Dd[i] = f2bf(S[i]);
}
__global__ void convw_kernel(const float* W, u16* Wt) {
    int i = blockIdx.x * blockDim.x + threadIdx.x;
    if (i >= D_ * D_ * 9) return;
    int kx = i % 3, ky = (i / 3) % 3, ci = (i / 9) % D_, o = i / (9 * D_);
    Wt[(long)o * (9 * D_) + (ky * 3 + kx) * D_ + ci] = f2bf(W[i]);
}

// FPN helpers ---------------------------------------------------------------
__global__ void chw2hwc_bf_kernel(const float* F, u16* A) {
    int idx = blockIdx.x * blockDim.x + threadIdx.x;
    if (idx >= BT_ * 9216 * D_) return;
    int c = idx & 255, pix = (idx >> 8) % 9216, bt = idx / (9216 * D_);
    A[idx] = f2bf(F[((long)(bt * D_) + c) * 9216 + pix]);
}

__global__ void fuse_up_kernel(const float* __restrict__ FU, const float* __restrict__ OUT,
                               u16* __restrict__ FB) {
    int idx = blockIdx.x * blockDim.x + threadIdx.x;
    if (idx >= BT_ * 9216 * D_) return;
    int c = idx & 255, pix = (idx >> 8) % 9216, bt = idx / (9216 * D_);
    int y = pix / 96, x = pix % 96;
    const long o0 = (long)BT_ * D_ * 9216;
    const float* m0 = OUT + o0 + ((long)(bt * D_) + c) * 2304;
    float sy = (y + 0.5f) * 0.5f - 0.5f, sx = (x + 0.5f) * 0.5f - 0.5f;
    float y0f = floorf(sy), x0f = floorf(sx);
    float wy = sy - y0f, wx = sx - x0f;
    int y0 = (int)y0f, x0 = (int)x0f;
    int y0c = min(max(y0, 0), 47), y1c = min(max(y0 + 1, 0), 47);
    int x0c = min(max(x0, 0), 47), x1c = min(max(x0 + 1, 0), 47);
    float v = m0[y0c * 48 + x0c] * (1 - wy) * (1 - wx) + m0[y0c * 48 + x1c] * (1 - wy) * wx
            + m0[y1c * 48 + x0c] * wy * (1 - wx)       + m0[y1c * 48 + x1c] * wy * wx;
    FB[idx] = f2bf(FU[idx] + v);
}

__global__ void gn_kernel(float* O, const float* gs, const float* gb) {
    __shared__ float r1[256], r2[256];
    int bt = blockIdx.x >> 5, g = blockIdx.x & 31;
    int t = threadIdx.x;
    const int CNT = 8 * 9216;
    float s = 0.f, sq = 0.f;
    for (int i = t; i < CNT; i += 256) {
        int c = g * 8 + i / 9216, pix = i % 9216;
        float v = O[((long)(bt * D_) + c) * 9216 + pix];
        s += v; sq += v * v;
    }
    r1[t] = s; r2[t] = sq; __syncthreads();
    for (int sh = 128; sh > 0; sh >>= 1) {
        if (t < sh) { r1[t] += r1[t + sh]; r2[t] += r2[t + sh]; }
        __syncthreads();
    }
    float mean = r1[0] / (float)CNT;
    float var  = r2[0] / (float)CNT - mean * mean;
    float rinv = rsqrtf(var + 1e-5f);
    for (int i = t; i < CNT; i += 256) {
        int c = g * 8 + i / 9216, pix = i % 9216;
        long id = ((long)(bt * D_) + c) * 9216 + pix;
        O[id] = (O[id] - mean) * rinv * gs[c] + gb[c];
    }
}

// ---------------------------------------------------------------------------
extern "C" void kernel_launch(void* const* d_in, const int* in_sizes, int n_in,
                              void* d_out, int out_size, void* d_ws, size_t ws_size,
                              hipStream_t stream) {
    (void)in_sizes; (void)n_in; (void)out_size; (void)ws_size;

    const float* feat_s4  = (const float*)d_in[0];
    const float* feat_s8  = (const float*)d_in[3];
    const float* pos_s8   = (const float*)d_in[4];
    const unsigned char* mask_s8 = (const unsigned char*)d_in[5];
    const float* feat_s16 = (const float*)d_in[6];
    const float* pos_s16  = (const float*)d_in[7];
    const unsigned char* mask_s16 = (const unsigned char*)d_in[8];
    const float* feat_s32 = (const float*)d_in[9];
    const float* pos_s32  = (const float*)d_in[10];
    const unsigned char* mask_s32 = (const unsigned char*)d_in[11];
    // params: fpn (adapt_W, adapt_b, conv_W, conv_b, gn_b, gn_s), layers, level_embed
    const float* adapt_W = (const float*)d_in[12];
    const float* adapt_b = (const float*)d_in[13];
    const float* conv_W  = (const float*)d_in[14];
    const float* conv_b  = (const float*)d_in[15];
    const float* gn_b    = (const float*)d_in[16];
    const float* gn_s    = (const float*)d_in[17];
    const int LB = 18;   // per-layer sorted: W1,W2,Wa,Wo,Wout,Wv,b1,b2,ba,bo,bout,bv,ln1_b,ln1_s,ln2_b,ln2_s
    const float* lvlemb  = (const float*)d_in[LB + 6 * 16];

    float* outp = (float*)d_out;

    // ---- workspace bump allocator -----------------------------------------
    char* ws = (char*)d_ws;
    size_t off = 0;
    auto alloc = [&](size_t bytes) -> void* {
        void* p = ws + off;
        off += (bytes + 255) & ~size_t(255);
        return p;
    };
    u16* Wo_t[6]; u16* Wa_t[6]; u16* Wv_t[6]; u16* Wout_t[6]; u16* W1_t[6]; u16* W2_t[6];
    for (int l = 0; l < 6; l++) {
        Wo_t[l]   = (u16*)alloc(192 * 256 * 2);
        Wa_t[l]   = (u16*)alloc(96 * 256 * 2);
        Wv_t[l]   = (u16*)alloc(256 * 256 * 2);
        Wout_t[l] = (u16*)alloc(256 * 256 * 2);
        W1_t[l]   = (u16*)alloc((size_t)2048 * 256 * 2);
        W2_t[l]   = (u16*)alloc((size_t)256 * 2048 * 2);
    }
    u16* adaptWt = (u16*)alloc(256 * 256 * 2);
    u16* convWt  = (u16*)alloc((size_t)256 * 2304 * 2);
    float* X  = (float*)alloc((size_t)M_ * D_ * 4);
    float* P  = (float*)alloc((size_t)M_ * D_ * 4);
    u16*   Xb = (u16*)alloc((size_t)M_ * D_ * 2);
    u16*   Qb = (u16*)alloc((size_t)M_ * D_ * 2);
    float* REF = (float*)alloc((size_t)BT_ * LEN_ * NL_ * 2 * 4);
    float* VR  = (float*)alloc(BT_ * NL_ * 2 * 4);
    unsigned char* MF = (unsigned char*)alloc(BT_ * LEN_);
    size_t scratch_off = off;
    float* Vv   = (float*)alloc((size_t)M_ * D_ * 4);
    float* OFFb = (float*)alloc((size_t)M_ * 192 * 4);
    float* ATb  = (float*)alloc((size_t)M_ * 96 * 4);
    u16*   MSb  = (u16*)alloc((size_t)M_ * D_ * 2);
    float* DLT  = (float*)alloc((size_t)M_ * D_ * 4);
    off = scratch_off;   // FPN aliases encoder scratch (encoder done by then)
    u16*   A4b    = (u16*)alloc((size_t)BT_ * 9216 * D_ * 2);
    float* FUSEDf = (float*)alloc((size_t)BT_ * 9216 * D_ * 4);
    u16*   FUSEDb = (u16*)alloc((size_t)BT_ * 9216 * D_ * 2);

    // ---- weight conversion -------------------------------------------------
    for (int l = 0; l < 6; l++) {
        const float* W1   = (const float*)d_in[LB + l * 16 + 0];
        const float* W2   = (const float*)d_in[LB + l * 16 + 1];
        const float* Wa   = (const float*)d_in[LB + l * 16 + 2];
        const float* Wo   = (const float*)d_in[LB + l * 16 + 3];
        const float* Wout = (const float*)d_in[LB + l * 16 + 4];
        const float* Wv   = (const float*)d_in[LB + l * 16 + 5];
        wt_transpose_kernel<<<(256 * 192 + 255) / 256, 256, 0, stream>>>(Wo, Wo_t[l], 256, 192);
        wt_transpose_kernel<<<(256 * 96 + 255) / 256, 256, 0, stream>>>(Wa, Wa_t[l], 256, 96);
        wt_transpose_kernel<<<(256 * 256 + 255) / 256, 256, 0, stream>>>(Wv, Wv_t[l], 256, 256);
        wt_transpose_kernel<<<(256 * 256 + 255) / 256, 256, 0, stream>>>(Wout, Wout_t[l], 256, 256);
        wt_transpose_kernel<<<(256 * 2048 + 255) / 256, 256, 0, stream>>>(W1, W1_t[l], 256, 2048);
        wt_transpose_kernel<<<(2048 * 256 + 255) / 256, 256, 0, stream>>>(W2, W2_t[l], 2048, 256);
    }
    cvt_kernel<<<(256 * 256 + 255) / 256, 256, 0, stream>>>(adapt_W, adaptWt, 256 * 256);
    convw_kernel<<<(256 * 256 * 9 + 255) / 256, 256, 0, stream>>>(conv_W, convWt);

    // ---- pack / reference points ------------------------------------------
    pack_kernel<<<(M_ * D_ + 255) / 256, 256, 0, stream>>>(
        feat_s8, feat_s16, feat_s32, pos_s8, pos_s16, pos_s32,
        mask_s8, mask_s16, mask_s32, lvlemb, X, P, Xb, Qb, MF);
    vr_kernel<<<1, 32, 0, stream>>>(mask_s8, mask_s16, mask_s32, VR);
    ref_kernel<<<(BT_ * LEN_ * NL_ + 255) / 256, 256, 0, stream>>>(VR, REF);

    // ---- encoder layers ----------------------------------------------------
    const int MT = M_ / 16;   // 1512
    for (int l = 0; l < 6; l++) {
        const float* b1   = (const float*)d_in[LB + l * 16 + 6];
        const float* b2   = (const float*)d_in[LB + l * 16 + 7];
        const float* ba   = (const float*)d_in[LB + l * 16 + 8];
        const float* bo   = (const float*)d_in[LB + l * 16 + 9];
        const float* bout = (const float*)d_in[LB + l * 16 + 10];
        const float* bv   = (const float*)d_in[LB + l * 16 + 11];
        const float* ln1b = (const float*)d_in[LB + l * 16 + 12];
        const float* ln1s = (const float*)d_in[LB + l * 16 + 13];
        const float* ln2b = (const float*)d_in[LB + l * 16 + 14];
        const float* ln2s = (const float*)d_in[LB + l * 16 + 15];

        gemm_bf16_kernel<4, 256, false><<<dim3(MT, 1), 128, 0, stream>>>(Qb, Wo_t[l], bo, OFFb, 192);
        gemm_bf16_kernel<2, 256, false><<<dim3(MT, 1), 128, 0, stream>>>(Qb, Wa_t[l], ba, ATb, 96);
        gemm_bf16_kernel<4, 256, false><<<dim3(MT, 1), 128, 0, stream>>>(Xb, Wv_t[l], bv, Vv, 256);
        mask_value_kernel<<<(M_ * D_ + 255) / 256, 256, 0, stream>>>(Vv, MF);
        softmax12_kernel<<<(M_ * NH_ + 255) / 256, 256, 0, stream>>>(ATb);
        ms_sample_kernel<<<(M_ * NH_ * 32 + 255) / 256, 256, 0, stream>>>(Vv, OFFb, ATb, REF, MSb);
        gemm_bf16_kernel<4, 256, false><<<dim3(MT, 1), 128, 0, stream>>>(MSb, Wout_t[l], bout, DLT, 256);
        add_ln_kernel<<<M_, 256, 0, stream>>>(X, DLT, ln1s, ln1b, Xb, nullptr, nullptr);
        ffn_kernel<<<MT, 256, 0, stream>>>(Xb, W1_t[l], b1, W2_t[l], b2, DLT);
        add_ln_kernel<<<M_, 256, 0, stream>>>(X, DLT, ln2s, ln2b, Xb, P, Qb);
    }

    // ---- outputs: mems, then FPN ------------------------------------------
    unflatten_kernel<<<(M_ * D_ + 255) / 256, 256, 0, stream>>>(X, outp);

    const int PIX = BT_ * 9216;
    chw2hwc_bf_kernel<<<(PIX * D_ + 255) / 256, 256, 0, stream>>>(feat_s4, A4b);
    gemm_bf16_kernel<4, 256, false><<<dim3(PIX / 16, 1), 128, 0, stream>>>(A4b, adaptWt, adapt_b, FUSEDf, 256);
    fuse_up_kernel<<<(PIX * D_ + 255) / 256, 256, 0, stream>>>(FUSEDf, outp, FUSEDb);
    conv3_kernel<<<dim3(BT_ * 96 * 6, 1), 128, 0, stream>>>(FUSEDb, convWt, conv_b, outp);
    gn_kernel<<<BT_ * 32, 256, 0, stream>>>(outp, gn_s, gn_b);
}
